// FourierCrossAttention_83614423318880
// MI455X (gfx1250) — matrix operand order repaired and checked by reading, not compiled, not run on hardware
//
#include <hip/hip_runtime.h>
#include <hip/hip_bf16.h>

typedef _Float16 h2  __attribute__((ext_vector_type(2)));
typedef _Float16 h8  __attribute__((ext_vector_type(8)));
typedef _Float16 h16 __attribute__((ext_vector_type(16)));
typedef float    f4  __attribute__((ext_vector_type(4)));
typedef float    f8  __attribute__((ext_vector_type(8)));

// Problem constants (from reference): B=8, L=4096, H=8, E=64, MODES=64
#define NL   4096
#define NE   64
#define NM   64
#define NBH  64      // B*H
#define MC   128     // 2*MODES (re rows 0..63, im rows 64..127)

static __device__ __forceinline__ h16 hcat(h8 a, h8 b) {
  return __builtin_shufflevector(a, b, 0,1,2,3,4,5,6,7,8,9,10,11,12,13,14,15);
}
static __device__ __forceinline__ f8 wmma_f16(h16 a, h16 b, f8 c) {
  // D = A(16x32 f16) * B(32x16 f16) + C(16x16 f32)
  return __builtin_amdgcn_wmma_f32_16x16x32_f16(false, a, false, b, (short)0, c, false, false);
}

// ---------------------------------------------------------------------------
// Kernel 1: precompute DFT bases in f16.
// F (forward): [2 tensors][128 rows][4096 t], row m = cos(2pi m t/L), row 64+m = -sin
// G (inverse): [4096 t][128 mc],  G[t][m] = c_m*cos, G[t][64+m] = -c_m*sin (c_0=1 else 2)
// ---------------------------------------------------------------------------
__global__ void __launch_bounds__(256) k_basis(const int* __restrict__ idxq,
                                               const int* __restrict__ idxkv,
                                               _Float16* __restrict__ F,
                                               _Float16* __restrict__ G) {
  int gid = blockIdx.x * 256 + threadIdx.x;   // 0 .. 128*4096-1
  int r = gid >> 12;                          // 0..127
  int t = gid & (NL - 1);
  int slot = r & (NM - 1);
  int is_sin = r >> 6;
  const float W = 6.28318530717958647692f / (float)NL;
  int mq = idxq[slot];
  int mk = idxkv[slot];
  float sq, cq, sk, ck;
  sincosf((float)((mq * t) & (NL - 1)) * W, &sq, &cq);
  sincosf((float)((mk * t) & (NL - 1)) * W, &sk, &ck);
  F[(size_t)r * NL + t]                     = (_Float16)(is_sin ? -sq : cq);
  F[(size_t)MC * NL + (size_t)r * NL + t]   = (_Float16)(is_sin ? -sk : ck);
  float cm = (mq == 0) ? 1.0f : 2.0f;
  G[(size_t)t * MC + r]                     = (_Float16)(is_sin ? -cm * sq : cm * cq);
}

// ---------------------------------------------------------------------------
// Stage one 32t x 64e chunk: f32 (t-major, stride 512) -> f16 transposed [e][32t].
// Thread owns t-pair (2p, 2p+1) x 4 consecutive e; packs pairs for b32 DS stores.
// ---------------------------------------------------------------------------
static __device__ __forceinline__ void stage_tile(const float* __restrict__ r0,
                                                  _Float16* sbuf, int e0, int p) {
  f4 x0 = *(const f4*)r0;
  f4 x1 = *(const f4*)(r0 + 512);
#pragma unroll
  for (int c = 0; c < 4; ++c) {
    h2 v = { (_Float16)x0[c], (_Float16)x1[c] };
    *(h2*)&sbuf[(e0 + c) * 32 + 2 * p] = v;
  }
}

// ---------------------------------------------------------------------------
// Kernel 2: forward DFT via WMMA.  One workgroup per (bh, tensor).
// Computes X[bh][mc=128][e=64] (f32) = F[128x4096] * slice[4096x64].
// 8 waves, wave w owns mode-strip rows 16w..16w+15 (4 chained 16x16 C tiles).
// Double-buffered LDS: stage chunk i+1 while WMMAs consume chunk i; one
// barrier per chunk.
// ---------------------------------------------------------------------------
__global__ void __launch_bounds__(256) k_fwd(const float* __restrict__ q,
                                             const float* __restrict__ kk,
                                             const _Float16* __restrict__ F,
                                             float* __restrict__ Xq,
                                             float* __restrict__ Xk) {
  const int bh = blockIdx.x, tensor = blockIdx.y;
  const float* src = tensor ? kk : q;
  const _Float16* Fb = F + (size_t)tensor * MC * NL;
  float* dst = (tensor ? Xk : Xq) + (size_t)bh * (MC * NE);
  const int b = bh >> 3, h = bh & 7;

  __shared__ _Float16 sB[2][NE * 32];       // double-buffered [e][t_local], 2 x 4 KB

  const int tid = threadIdx.x;
  const int lane = tid & 31, w = tid >> 5;
  const int l16 = lane & 15, hi = lane >> 4;
  const int p = tid & 15;                   // t-pair index within 32-chunk
  const int e0 = (tid >> 4) * 4;            // 4 consecutive e per thread

  f8 acc0 = {0,0,0,0,0,0,0,0}, acc1 = acc0, acc2 = acc0, acc3 = acc0;

  // base pointer for this thread's staging rows (t = kb + 2p, stride H*E = 512 floats)
  const float* gbase = src + ((size_t)(b * NL + 2 * p) * 8 + h) * NE + e0;

  stage_tile(gbase, sB[0], e0, p);
  __syncthreads();

  for (int kb = 0; kb < NL; kb += 32) {
    const int cur = (kb >> 5) & 1;
    // ---- stage next chunk into the other buffer (no barrier needed yet) ----
    if (kb + 32 < NL) {
      stage_tile(gbase + (size_t)(kb + 32) * 512, sB[cur ^ 1], e0, p);
      if (kb + 64 < NL) __builtin_prefetch(gbase + (size_t)(kb + 64) * 512, 0, 3);
    }

    // ---- A tile: basis rows 16w..16w+15, K = kb..kb+31, WMMA A-layout ----
    const _Float16* pA = Fb + (size_t)(16 * w + l16) * NL + kb;
    h16 A = hcat(*(const h8*)(pA + hi * 16), *(const h8*)(pA + hi * 16 + 8));

    // ---- B tiles from LDS (column e = n*16 + l16, 16 contiguous t) ----
    const _Float16* pB = &sB[cur][(size_t)l16 * 32 + hi * 16];
    h16 B0 = hcat(*(const h8*)(pB +    0), *(const h8*)(pB +    8));
    h16 B1 = hcat(*(const h8*)(pB +  512), *(const h8*)(pB +  520));
    h16 B2 = hcat(*(const h8*)(pB + 1024), *(const h8*)(pB + 1032));
    h16 B3 = hcat(*(const h8*)(pB + 1536), *(const h8*)(pB + 1544));

    acc0 = wmma_f16(A, B0, acc0);
    acc1 = wmma_f16(A, B1, acc1);
    acc2 = wmma_f16(A, B2, acc2);
    acc3 = wmma_f16(A, B3, acc3);

    // One barrier per chunk: orders this chunk's reads against next iteration's
    // writes into this buffer, and the just-issued writes against next reads.
    __syncthreads();
  }

  // ---- store C tiles: element (v, lane) -> M = v + 8*hi, N = l16 ----
#pragma unroll
  for (int v = 0; v < 8; ++v) {
    int row = 16 * w + v + 8 * hi;
    dst[row * NE +  0 + l16] = acc0[v];
    dst[row * NE + 16 + l16] = acc1[v];
    dst[row * NE + 32 + l16] = acc2[v];
    dst[row * NE + 48 + l16] = acc3[v];
  }
}

// ---------------------------------------------------------------------------
// Kernel 3: middle stage 1 (per bh): xqk = Xq^T Xk (complex, over e), tanh on
// Re/Im, then xqkv[e][x] = sum_y xqk[x][y]*Xk[e][y]  -> XV (f32, global ws).
// ---------------------------------------------------------------------------
__global__ void __launch_bounds__(256) k_mid1(const float* __restrict__ Xq,
                                              const float* __restrict__ Xk,
                                              float* __restrict__ XV) {
  const int bh = blockIdx.x;
  __shared__ float sXk[MC * NE];     // 32 KB: [mc][e]
  __shared__ float sQK[2 * NM * NM]; // 32 KB: [re/im][x][y]
  const int tid = threadIdx.x;
  const float* xkg = Xk + (size_t)bh * (MC * NE);
  const float* xqg = Xq + (size_t)bh * (MC * NE);
#pragma unroll
  for (int i = 0; i < 32; ++i) sXk[tid + i * 256] = xkg[tid + i * 256];
  __syncthreads();

  { // xqk[x][y] = sum_e Xq[e,x]*Xk[e,y]  (no conjugate), then tanh per part
    const int x = tid & 63, y0 = (tid >> 6) * 16;
    float aR[16] = {}, aI[16] = {};
    for (int e = 0; e < NE; ++e) {
      float qr = xqg[x * NE + e], qi = xqg[(NM + x) * NE + e];
#pragma unroll
      for (int j = 0; j < 16; ++j) {
        float kr = sXk[(y0 + j) * NE + e], ki = sXk[(NM + y0 + j) * NE + e];
        aR[j] += qr * kr - qi * ki;
        aI[j] += qr * ki + qi * kr;
      }
    }
#pragma unroll
    for (int j = 0; j < 16; ++j) {
      sQK[x * NM + y0 + j]        = tanhf(aR[j]);
      sQK[4096 + x * NM + y0 + j] = tanhf(aI[j]);
    }
  }
  __syncthreads();

  { // xqkv[e][x] = sum_y xqk[x][y]*Xk[e][y]
    const int e = tid & 63, x0 = (tid >> 6) * 16;
    float aR[16] = {}, aI[16] = {};
    for (int y = 0; y < NM; ++y) {
      float kr = sXk[y * NE + e], ki = sXk[(NM + y) * NE + e];
#pragma unroll
      for (int j = 0; j < 16; ++j) {
        float qkR = sQK[(x0 + j) * NM + y], qkI = sQK[4096 + (x0 + j) * NM + y];
        aR[j] += qkR * kr - qkI * ki;
        aI[j] += qkR * ki + qkI * kr;
      }
    }
    float* xv = XV + (size_t)bh * 8192;
#pragma unroll
    for (int j = 0; j < 16; ++j) {
      xv[e * NM + x0 + j]        = aR[j];
      xv[4096 + e * NM + x0 + j] = aI[j];
    }
  }
}

// ---------------------------------------------------------------------------
// Kernel 4: middle stage 2 (per bh): xqkvw[o][x] = sum_e xqkv[e][x]*w[h,e,o,x]
// (complex weights w1 + i*w2). Emits Y in f16, WMMA-A-friendly [o][128] rows.
// ---------------------------------------------------------------------------
__global__ void __launch_bounds__(256) k_mid2(const float* __restrict__ XV,
                                              const float* __restrict__ W1,
                                              const float* __restrict__ W2,
                                              _Float16* __restrict__ Y) {
  const int bh = blockIdx.x, h = bh & 7;
  __shared__ float sV[8192];         // [re/im][e][x]
  const int tid = threadIdx.x;
  const float* xv = XV + (size_t)bh * 8192;
#pragma unroll
  for (int i = 0; i < 32; ++i) sV[tid + i * 256] = xv[tid + i * 256];
  __syncthreads();

  const int x = tid & 63, o0 = (tid >> 6) * 16;
  float aR[16] = {}, aI[16] = {};
  const float* w1h = W1 + (size_t)h * NE * NE * NM;
  const float* w2h = W2 + (size_t)h * NE * NE * NM;
  for (int e = 0; e < NE; ++e) {
    float vr = sV[e * NM + x], vi = sV[4096 + e * NM + x];
    const float* w1e = w1h + (size_t)e * (NE * NM) + x;
    const float* w2e = w2h + (size_t)e * (NE * NM) + x;
#pragma unroll
    for (int j = 0; j < 16; ++j) {
      float w1v = w1e[(o0 + j) * NM];
      float w2v = w2e[(o0 + j) * NM];
      aR[j] += vr * w1v - vi * w2v;
      aI[j] += vr * w2v + vi * w1v;
    }
  }
  _Float16* yb = Y + (size_t)bh * NE * MC;
#pragma unroll
  for (int j = 0; j < 16; ++j) {
    yb[(o0 + j) * MC + x]      = (_Float16)aR[j];   // Re coefficient slot x
    yb[(o0 + j) * MC + NM + x] = (_Float16)aI[j];   // Im coefficient slot x
  }
}

// ---------------------------------------------------------------------------
// Kernel 5: inverse rDFT via WMMA. out[bh][o][t] = scale * sum_mc Y[o][mc]*G[t][mc]
// Grid (8 t-blocks of 512, 64 bh). Wave: o-strip (w&3)*16, t-half (w>>2).
// Y strip held in 4 resident A tiles; G streamed as B tiles from global (L2).
// ---------------------------------------------------------------------------
__global__ void __launch_bounds__(256) k_inv(const _Float16* __restrict__ Y,
                                             const _Float16* __restrict__ G,
                                             float* __restrict__ out) {
  const int bh = blockIdx.y;
  const int tbase = blockIdx.x * 512;
  const int tid = threadIdx.x, lane = tid & 31, w = tid >> 5;
  const int l16 = lane & 15, hi = lane >> 4;
  const int o0 = 16 * (w & 3), thalf = w >> 2;

  h16 Areg[4];
  const _Float16* pY = Y + ((size_t)bh * NE + o0 + l16) * MC;
#pragma unroll
  for (int c = 0; c < 4; ++c) {
    const _Float16* pa = pY + 32 * c;
    Areg[c] = hcat(*(const h8*)(pa + hi * 16), *(const h8*)(pa + hi * 16 + 8));
  }

  const float SCALE = 1.0f / 1073741824.0f;  // 1/(IN_CH*OUT_CH*L) = 2^-30
#pragma unroll 1
  for (int j = 0; j < 16; ++j) {
    const int t0 = tbase + thalf * 256 + j * 16;
    f8 acc = {0,0,0,0,0,0,0,0};
#pragma unroll
    for (int c = 0; c < 4; ++c) {
      const _Float16* pG = G + (size_t)(t0 + l16) * MC + 32 * c + hi * 16;
      h16 Bv = hcat(*(const h8*)pG, *(const h8*)(pG + 8));
      acc = wmma_f16(Areg[c], Bv, acc);
    }
    float* po = out + ((size_t)bh * NE + o0) * NL + t0 + l16;
#pragma unroll
    for (int v = 0; v < 8; ++v) po[(size_t)(v + 8 * hi) * NL] = acc[v] * SCALE;
  }
}

// ---------------------------------------------------------------------------
extern "C" void kernel_launch(void* const* d_in, const int* in_sizes, int n_in,
                              void* d_out, int out_size, void* d_ws, size_t ws_size,
                              hipStream_t stream) {
  const float* q   = (const float*)d_in[0];
  const float* k   = (const float*)d_in[1];
  // d_in[2] == v : unused by the reference computation
  const float* w1  = (const float*)d_in[3];
  const float* w2  = (const float*)d_in[4];
  const int* idxq  = (const int*)d_in[5];
  const int* idxkv = (const int*)d_in[6];
  float* outp = (float*)d_out;

  char* ws = (char*)d_ws;
  _Float16* F  = (_Float16*)(ws + 0);                    // 2*128*4096 f16 = 2 MB
  _Float16* G  = (_Float16*)(ws + 2 * 1048576);          // 4096*128 f16  = 1 MB
  float*    Xq = (float*)   (ws + 3 * 1048576);          // 64*128*64 f32 = 2 MB
  float*    Xk = (float*)   (ws + 5 * 1048576);          // 2 MB
  float*    XV = (float*)   (ws + 7 * 1048576);          // 2 MB
  _Float16* Y  = (_Float16*)(ws + 9 * 1048576);          // 64*64*128 f16 = 1 MB
  // total ws use: 10 MB

  k_basis<<<dim3(2048), dim3(256), 0, stream>>>(idxq, idxkv, F, G);
  k_fwd  <<<dim3(64, 2), dim3(256), 0, stream>>>(q, k, F, Xq, Xk);
  k_mid1 <<<dim3(64), dim3(256), 0, stream>>>(Xq, Xk, XV);
  k_mid2 <<<dim3(64), dim3(256), 0, stream>>>(XV, w1, w2, Y);
  k_inv  <<<dim3(8, 64), dim3(256), 0, stream>>>(Y, G, outp);
}
